// CrystallGNN_46042049413576
// MI455X (gfx1250) — matrix operand order repaired and checked by reading, not compiled, not run on hardware
//
#include <hip/hip_runtime.h>
#include <hip/hip_bf16.h>

// ---------------------------------------------------------------------------
// CrystalGNN (SchNet-style) for MI455X / gfx1250, wave32, WMMA bf16->f32.
// ---------------------------------------------------------------------------

#define N_ATOM_FEATS 64
#define N_RBF        10
#define N_CONV       3
#define N_HIDDEN     64
#define N_NODES      50000
#define N_EDGES      800000
#define N_GRAPHS     512

typedef __attribute__((ext_vector_type(16))) __bf16 v16bf;
typedef __attribute__((ext_vector_type(8)))  float  v8f;

// ---- workspace layout (bytes) ----
#define OFF_X      0u                    // 50000*64 f32  = 12,800,000 B
#define OFF_H      12800000u             // 12,800,000 B
#define OFF_AGG    25600000u             // 12,800,000 B
#define OFF_SUMS   38400000u             // 512*64 f32    = 131,072 B
#define OFF_CNTS   38531072u             // 512 f32       = 2,048 B
#define OFF_C      38533120u             // 512*128 f32   = 262,144 B
#define OFF_W1BF   38795264u             // 3*4096 bf16   = 24,576 B (32B aligned)
#define OFF_W2BF   38819840u             // 3*4096 bf16   = 24,576 B
#define OFF_WSBF   38844416u             // 8192 bf16     = 16,384 B

__device__ __forceinline__ float softplus_f(float v) {
    // numerically-stable softplus: max(v,0) + log1p(exp(-|v|))
    return fmaxf(v, 0.0f) + log1pf(__expf(-fabsf(v)));
}

// ---------------------------------------------------------------------------
// Embedding gather: x[n,f] = emb[x_ids[n], f]
// ---------------------------------------------------------------------------
__global__ void k_embed(const int* __restrict__ ids, const float* __restrict__ emb,
                        float* __restrict__ x) {
    int idx = blockIdx.x * blockDim.x + threadIdx.x;
    if (idx >= N_NODES * 64) return;
    int n = idx >> 6, f = idx & 63;
    x[idx] = emb[ids[n] * 64 + f];
}

// ---------------------------------------------------------------------------
// Swizzle a row-major [K,N] f32 weight into bf16 WMMA B-fragments.
// Fragment f = kb*(N/16)+nt; element layout per CDNA5 ISA:
//   lane L holds column n = nt*16 + (L&15); values t=0..15 cover
//   k = kb*32 + (t>>3)*16 + (L>>4)*8 + (t&7).  One v16bf (32B) per lane, contiguous.
// ---------------------------------------------------------------------------
__global__ void k_swizzle_w(const float* __restrict__ W, __bf16* __restrict__ dst,
                            int N, int total) {
    int idx = blockIdx.x * blockDim.x + threadIdx.x;
    if (idx >= total) return;
    int f    = idx >> 9;
    int rest = idx & 511;
    int lane = rest >> 4;
    int t    = rest & 15;
    int ntiles = N >> 4;
    int kb = f / ntiles;
    int nt = f % ntiles;
    int k  = kb * 32 + (t >> 3) * 16 + (lane >> 4) * 8 + (t & 7);
    int n  = nt * 16 + (lane & 15);
    dst[idx] = (__bf16)W[k * N + n];
}

// ---------------------------------------------------------------------------
// Node GEMM: out[M,64] = act( in[M,64] @ W(64x64, swizzled bf16) + bias (+resid) )
// One wave per 16-row M-tile; 4 acc frags (N=64), K=64 in two 32-wide steps.
// Epilogue transposes the C fragments through a padded LDS tile so the
// residual loads and output stores are fully-coalesced b128 accesses.
// ---------------------------------------------------------------------------
#define TILE_STRIDE 68   // 16-row tile, row stride 68 words: bank-conflict-free
__global__ void __launch_bounds__(128)
k_node_gemm(const float* __restrict__ in, const __bf16* __restrict__ Wswz,
            const float* __restrict__ bias, const float* __restrict__ resid,
            float* __restrict__ out, int act, int mtiles) {
    __shared__ __align__(16) float tile[4][16 * TILE_STRIDE];
    const int wave = threadIdx.x >> 5;
    const int lane = threadIdx.x & 31;
    const int mtile = blockIdx.x * 4 + wave;
    if (mtile >= mtiles) return;                 // wave-uniform: EXEC stays all-1s

    const int mrow = (mtile << 4) + (lane & 15); // A-row owned by this lane
    const int base = (lane >> 4) << 3;           // 0 or 8 (k sub-offset)

    v8f acc[4] = {v8f{}, v8f{}, v8f{}, v8f{}};

#pragma unroll
    for (int kb = 0; kb < 2; ++kb) {
        const float* rowp = in + (size_t)mrow * 64 + kb * 32;
        float4 p0 = *(const float4*)(rowp + base);
        float4 p1 = *(const float4*)(rowp + base + 4);
        float4 p2 = *(const float4*)(rowp + 16 + base);
        float4 p3 = *(const float4*)(rowp + 16 + base + 4);
        v16bf a;
        a[0] = (__bf16)p0.x;  a[1] = (__bf16)p0.y;  a[2]  = (__bf16)p0.z;  a[3]  = (__bf16)p0.w;
        a[4] = (__bf16)p1.x;  a[5] = (__bf16)p1.y;  a[6]  = (__bf16)p1.z;  a[7]  = (__bf16)p1.w;
        a[8] = (__bf16)p2.x;  a[9] = (__bf16)p2.y;  a[10] = (__bf16)p2.z;  a[11] = (__bf16)p2.w;
        a[12] = (__bf16)p3.x; a[13] = (__bf16)p3.y; a[14] = (__bf16)p3.z;  a[15] = (__bf16)p3.w;

#pragma unroll
        for (int nt = 0; nt < 4; ++nt) {
            v16bf b = *(const v16bf*)(Wswz + ((size_t)(kb * 4 + nt) * 32 + lane) * 16);
            acc[nt] = __builtin_amdgcn_wmma_f32_16x16x32_bf16(
                false, a, false, b, (short)0, acc[nt], false, false);
        }
    }

    // --- epilogue: transpose C frags via LDS (wave-private tile) ---
    float* st = tile[wave];
    const int ncol  = lane & 15;
    const int rbase = (lane >> 4) << 3;
#pragma unroll
    for (int nt = 0; nt < 4; ++nt) {
        float bv = bias[nt * 16 + ncol];
#pragma unroll
        for (int r = 0; r < 8; ++r)
            st[(rbase + r) * TILE_STRIDE + nt * 16 + ncol] = acc[nt][r] + bv;
    }
    __builtin_amdgcn_wave_barrier();   // LDS same-wave ops complete in-order (ISA)

    const int row = lane >> 1;
    const int cb  = (lane & 1) * 32;
    const float* sr = st + row * TILE_STRIDE + cb;
    const int grow = (mtile << 4) + row;
    float* op = out + (size_t)grow * 64 + cb;
    const float* rp = resid ? resid + (size_t)grow * 64 + cb : (const float*)0;
#pragma unroll
    for (int j = 0; j < 8; ++j) {
        float4 v = *(const float4*)(sr + 4 * j);
        if (rp) {
            float4 rv = *(const float4*)(rp + 4 * j);
            v.x += rv.x; v.y += rv.y; v.z += rv.z; v.w += rv.w;
        }
        if (act) {
            v.x = softplus_f(v.x); v.y = softplus_f(v.y);
            v.z = softplus_f(v.z); v.w = softplus_f(v.w);
        }
        *(float4*)(op + 4 * j) = v;
    }
}

// ---------------------------------------------------------------------------
// Edge messages: per edge e, f = rbf(d) @ We + be ; m = h[src]*f ; agg[dst] += m
// 32 edges per 256-thread block, 8 threads/edge, 8 feats/thread.
// We/be staged in LDS. RBF recomputed inline (saves 32MB x 4 of traffic).
// h/agg are 12.8MB each -> L2-resident gather/scatter.
// ---------------------------------------------------------------------------
__global__ void __launch_bounds__(256)
k_message(const int* __restrict__ ei, const float* __restrict__ ea,
          const float* __restrict__ We, const float* __restrict__ be,
          const float* __restrict__ h, float* __restrict__ agg) {
    __shared__ float sWe[N_RBF * 64];
    __shared__ float sbe[64];
    int tid = threadIdx.x;
    for (int i = tid; i < N_RBF * 64; i += 256) sWe[i] = We[i];
    if (tid < 64) sbe[tid] = be[tid];
    __syncthreads();

    int eid = blockIdx.x * 32 + (tid >> 3);
    if (eid >= N_EDGES) return;

    int   src = ei[eid];
    int   dst = ei[N_EDGES + eid];
    float d   = ea[eid];

    float rbf[N_RBF];
#pragma unroll
    for (int r = 0; r < N_RBF; ++r) {
        float t = d - (float)r * (6.0f / 9.0f);
        rbf[r] = __expf(-1.125f * t * t);   // coeff = -0.5/(6/9)^2
    }

    int fb = (tid & 7) * 8;
    const float* hp = h + (size_t)src * 64 + fb;
    float4 h0 = *(const float4*)(hp);
    float4 h1 = *(const float4*)(hp + 4);
    float hv[8] = {h0.x, h0.y, h0.z, h0.w, h1.x, h1.y, h1.z, h1.w};

    float* ap = agg + (size_t)dst * 64 + fb;
#pragma unroll
    for (int u = 0; u < 8; ++u) {
        float f = sbe[fb + u];
#pragma unroll
        for (int r = 0; r < N_RBF; ++r) f = fmaf(rbf[r], sWe[r * 64 + fb + u], f);
        atomicAdd(&ap[u], hv[u] * f);
    }
}

// ---------------------------------------------------------------------------
// Global mean-pool accumulation (sums + counts via f32 atomics, L2-resident)
// ---------------------------------------------------------------------------
__global__ void k_pool(const float* __restrict__ x, const int* __restrict__ batch,
                       float* __restrict__ sums, float* __restrict__ cnts) {
    int idx = blockIdx.x * blockDim.x + threadIdx.x;
    if (idx >= N_NODES * 64) return;
    int n = idx >> 6, f = idx & 63;
    int g = batch[n];
    atomicAdd(&sums[g * 64 + f], x[idx]);
    if (f == 0) atomicAdd(&cnts[g], 1.0f);
}

// ---------------------------------------------------------------------------
// Head GEMM: c[512,128] = relu( mean[512,64] @ Ws(64x128, swizzled) + bs )
// One wave per 16-row M-tile (32 tiles), 8 N-tiles, K=64.
// ---------------------------------------------------------------------------
__global__ void __launch_bounds__(128)
k_head_gemm(const float* __restrict__ sums, const float* __restrict__ cnts,
            const __bf16* __restrict__ Wswz, const float* __restrict__ bs,
            float* __restrict__ c) {
    const int wave = threadIdx.x >> 5;
    const int lane = threadIdx.x & 31;
    const int mtile = blockIdx.x * 4 + wave;   // grid=8 -> 32 tiles == 512 rows
    const int mrow = (mtile << 4) + (lane & 15);
    const int base = (lane >> 4) << 3;
    const float rc = 1.0f / fmaxf(cnts[mrow], 1.0f);

    v8f acc[8] = {v8f{}, v8f{}, v8f{}, v8f{}, v8f{}, v8f{}, v8f{}, v8f{}};

#pragma unroll
    for (int kb = 0; kb < 2; ++kb) {
        const float* rowp = sums + (size_t)mrow * 64 + kb * 32;
        float4 p0 = *(const float4*)(rowp + base);
        float4 p1 = *(const float4*)(rowp + base + 4);
        float4 p2 = *(const float4*)(rowp + 16 + base);
        float4 p3 = *(const float4*)(rowp + 16 + base + 4);
        v16bf a;
        a[0] = (__bf16)(p0.x * rc);  a[1] = (__bf16)(p0.y * rc);
        a[2] = (__bf16)(p0.z * rc);  a[3] = (__bf16)(p0.w * rc);
        a[4] = (__bf16)(p1.x * rc);  a[5] = (__bf16)(p1.y * rc);
        a[6] = (__bf16)(p1.z * rc);  a[7] = (__bf16)(p1.w * rc);
        a[8] = (__bf16)(p2.x * rc);  a[9] = (__bf16)(p2.y * rc);
        a[10] = (__bf16)(p2.z * rc); a[11] = (__bf16)(p2.w * rc);
        a[12] = (__bf16)(p3.x * rc); a[13] = (__bf16)(p3.y * rc);
        a[14] = (__bf16)(p3.z * rc); a[15] = (__bf16)(p3.w * rc);

#pragma unroll
        for (int nt = 0; nt < 8; ++nt) {
            v16bf b = *(const v16bf*)(Wswz + ((size_t)(kb * 8 + nt) * 32 + lane) * 16);
            acc[nt] = __builtin_amdgcn_wmma_f32_16x16x32_bf16(
                false, a, false, b, (short)0, acc[nt], false, false);
        }
    }

    const int ncol  = lane & 15;
    const int rbase = (lane >> 4) << 3;
#pragma unroll
    for (int nt = 0; nt < 8; ++nt) {
        float bv = bs[nt * 16 + ncol];
#pragma unroll
        for (int r = 0; r < 8; ++r) {
            int row = (mtile << 4) + rbase + r;
            float v = fmaxf(acc[nt][r] + bv, 0.0f);
            c[row * 128 + nt * 16 + ncol] = v;
        }
    }
}

// ---------------------------------------------------------------------------
// Output heads: out_bg[g] = relu(c[g]@Wbg1+bbg1)@Wbg2+bbg2 (and eh head).
// One 64-thread block per graph; LDS tree reduction.
// ---------------------------------------------------------------------------
__global__ void __launch_bounds__(64)
k_heads(const float* __restrict__ c,
        const float* __restrict__ Wbg1, const float* __restrict__ bbg1,
        const float* __restrict__ Wbg2, const float* __restrict__ bbg2,
        const float* __restrict__ Weh1, const float* __restrict__ beh1,
        const float* __restrict__ Weh2, const float* __restrict__ beh2,
        float* __restrict__ out) {
    __shared__ float sb[64];
    __shared__ float se[64];
    int g = blockIdx.x, j = threadIdx.x;
    const float* cg = c + (size_t)g * 128;
    float hb = bbg1[j], he = beh1[j];
    for (int r = 0; r < 128; ++r) {
        float cv = cg[r];
        hb = fmaf(cv, Wbg1[r * 64 + j], hb);
        he = fmaf(cv, Weh1[r * 64 + j], he);
    }
    sb[j] = fmaxf(hb, 0.0f) * Wbg2[j];
    se[j] = fmaxf(he, 0.0f) * Weh2[j];
    __syncthreads();
    for (int s = 32; s > 0; s >>= 1) {
        if (j < s) { sb[j] += sb[j + s]; se[j] += se[j + s]; }
        __syncthreads();
    }
    if (j == 0) {
        out[g]             = sb[0] + bbg2[0];
        out[N_GRAPHS + g]  = se[0] + beh2[0];
    }
}

// ---------------------------------------------------------------------------
extern "C" void kernel_launch(void* const* d_in, const int* in_sizes, int n_in,
                              void* d_out, int out_size, void* d_ws, size_t ws_size,
                              hipStream_t stream) {
    (void)in_sizes; (void)n_in; (void)out_size; (void)ws_size;

    const int*   x_ids   = (const int*)d_in[0];
    const int*   eindex  = (const int*)d_in[1];
    const float* eattr   = (const float*)d_in[2];
    const int*   batch   = (const int*)d_in[3];
    const float* emb     = (const float*)d_in[4];
    const float* blk_W1  = (const float*)d_in[5];
    const float* blk_b1  = (const float*)d_in[6];
    const float* blk_We  = (const float*)d_in[7];
    const float* blk_be  = (const float*)d_in[8];
    const float* blk_W2  = (const float*)d_in[9];
    const float* blk_b2  = (const float*)d_in[10];
    const float* Ws      = (const float*)d_in[11];
    const float* bs      = (const float*)d_in[12];
    const float* Wbg1    = (const float*)d_in[13];
    const float* bbg1    = (const float*)d_in[14];
    const float* Wbg2    = (const float*)d_in[15];
    const float* bbg2    = (const float*)d_in[16];
    const float* Weh1    = (const float*)d_in[17];
    const float* beh1    = (const float*)d_in[18];
    const float* Weh2    = (const float*)d_in[19];
    const float* beh2    = (const float*)d_in[20];

    char*   wsb  = (char*)d_ws;
    float*  x    = (float*)(wsb + OFF_X);
    float*  h    = (float*)(wsb + OFF_H);
    float*  agg  = (float*)(wsb + OFF_AGG);
    float*  sums = (float*)(wsb + OFF_SUMS);
    float*  cnts = (float*)(wsb + OFF_CNTS);
    float*  c    = (float*)(wsb + OFF_C);
    __bf16* w1bf = (__bf16*)(wsb + OFF_W1BF);
    __bf16* w2bf = (__bf16*)(wsb + OFF_W2BF);
    __bf16* wsbf = (__bf16*)(wsb + OFF_WSBF);

    float* out = (float*)d_out;

    // --- embedding lookup ---
    k_embed<<<(N_NODES * 64 + 255) / 256, 256, 0, stream>>>(x_ids, emb, x);

    // --- swizzle weights to WMMA B-fragment bf16 layout ---
    for (int i = 0; i < N_CONV; ++i) {
        k_swizzle_w<<<(8 * 512 + 255) / 256, 256, 0, stream>>>(
            blk_W1 + (size_t)i * 4096, w1bf + (size_t)i * 4096, 64, 8 * 512);
        k_swizzle_w<<<(8 * 512 + 255) / 256, 256, 0, stream>>>(
            blk_W2 + (size_t)i * 4096, w2bf + (size_t)i * 4096, 64, 8 * 512);
    }
    k_swizzle_w<<<(16 * 512 + 255) / 256, 256, 0, stream>>>(Ws, wsbf, 128, 16 * 512);

    const int mtiles = N_NODES / 16;                 // 3125
    const int gemm_grid = (mtiles + 3) / 4;          // 782 blocks x 4 waves

    // --- interaction blocks ---
    for (int i = 0; i < N_CONV; ++i) {
        k_node_gemm<<<gemm_grid, 128, 0, stream>>>(
            x, w1bf + (size_t)i * 4096, blk_b1 + (size_t)i * 64,
            nullptr, h, /*act=*/0, mtiles);

        hipMemsetAsync(agg, 0, (size_t)N_NODES * 64 * sizeof(float), stream);

        k_message<<<N_EDGES / 32, 256, 0, stream>>>(
            eindex, eattr, blk_We + (size_t)i * N_RBF * 64,
            blk_be + (size_t)i * 64, h, agg);

        k_node_gemm<<<gemm_grid, 128, 0, stream>>>(
            agg, w2bf + (size_t)i * 4096, blk_b2 + (size_t)i * 64,
            /*resid=*/x, /*out=*/x, /*act(softplus)=*/1, mtiles);
    }

    // --- global mean pool ---
    hipMemsetAsync(sums, 0, (size_t)(N_GRAPHS * 64 + N_GRAPHS) * sizeof(float), stream);
    k_pool<<<(N_NODES * 64 + 255) / 256, 256, 0, stream>>>(x, batch, sums, cnts);

    // --- shared head GEMM: c = relu(mean @ Ws + bs) ---
    k_head_gemm<<<8, 128, 0, stream>>>(sums, cnts, wsbf, bs, c);

    // --- two output heads ---
    k_heads<<<N_GRAPHS, 64, 0, stream>>>(c, Wbg1, bbg1, Wbg2, bbg2,
                                         Weh1, beh1, Weh2, beh2, out);
}
